// MultiHeadAttention_20461224198307
// MI455X (gfx1250) — matrix operand order repaired and checked by reading, not compiled, not run on hardware
//
#include <hip/hip_runtime.h>
#include <math.h>

typedef __bf16 bf16_t;
typedef __attribute__((ext_vector_type(4)))  float   v4f;
typedef __attribute__((ext_vector_type(8)))  float   v8f;
typedef __attribute__((ext_vector_type(4)))  __bf16  v4bf;
typedef __attribute__((ext_vector_type(8)))  __bf16  v8bf;
typedef __attribute__((ext_vector_type(16))) __bf16  v16bf;

#define EMB   768
#define NHEAD 8
#define DH    96
#define NSEQ  2048
#define MTOT  8192   // B * NSEQ

// ---------- helpers ----------

__device__ __forceinline__ v8f vzero8() {
  v8f z;
#pragma unroll
  for (int i = 0; i < 8; ++i) z[i] = 0.f;
  return z;
}

// A-operand fragment (16x32 bf16): lane l(0-15): M=l, K={0..7,16..23}; lane l+16: M=l, K={8..15,24..31}.
// Caller passes p = &row[kbase + 8*h]; we load 8 bf16 at p and 8 at p+16.
__device__ __forceinline__ v16bf load_a_frag(const bf16_t* __restrict__ p) {
  v8bf lo = *(const v8bf*)p;
  v8bf hi = *(const v8bf*)(p + 16);
  return __builtin_shufflevector(lo, hi, 0,1,2,3,4,5,6,7,8,9,10,11,12,13,14,15);
}

// B-operand fragment (32x16 bf16): lane l(0-15): N=l, K=0..15 contiguous; lane l+16: N=l, K=16..31.
// Caller passes p = &row[kbase + 16*h]; 32B aligned contiguous load.
__device__ __forceinline__ v16bf load_b_frag(const bf16_t* __restrict__ p) {
  return *(const v16bf*)p;
}

__device__ __forceinline__ v8f wmma_bf16(v16bf a, v16bf b, v8f c) {
  return __builtin_amdgcn_wmma_f32_16x16x32_bf16(false, a, false, b, (short)0, c, false, false);
}

// ---------- fp32 -> bf16 conversion (vectorized x4) ----------

__global__ __launch_bounds__(256) void cvt_kernel(const float* __restrict__ in,
                                                  bf16_t* __restrict__ out, int n4) {
  int i = blockIdx.x * blockDim.x + threadIdx.x;
  if (i < n4) {
    v4f v = ((const v4f*)in)[i];
    v4bf o;
#pragma unroll
    for (int e = 0; e < 4; ++e) o[e] = (bf16_t)v[e];
    ((v4bf*)out)[i] = o;
  }
}

// ---------- projection GEMM: C = A[M,768]bf16 * W[768,768]bf16^T + bias ----------
// MODE 0: store fp32 row-major [M,768]           (final output projection)
// MODE 1: store bf16, scaled, to [bh][NSEQ][DH]  (Q / K)
// MODE 2: store bf16, scaled, to [bh][DH][NSEQ]  (V transposed)
template<int MODE>
__global__ __launch_bounds__(256) void gemm_kernel(const bf16_t* __restrict__ A,
                                                   const bf16_t* __restrict__ W,
                                                   const float* __restrict__ bias,
                                                   void* __restrict__ outp,
                                                   float out_scale) {
  const int lane = threadIdx.x & 31;
  const int wv   = threadIdx.x >> 5;
  const int l = lane & 15, h = lane >> 4;
  const int wm = wv & 3, wn = wv >> 2;                 // 4x2 waves of 32x64 each
  const int row0 = blockIdx.y * 128 + wm * 32;
  const int col0 = blockIdx.x * 128 + wn * 64;

  v8f acc[2][4];
#pragma unroll
  for (int mt = 0; mt < 2; ++mt)
#pragma unroll
    for (int nt = 0; nt < 4; ++nt) acc[mt][nt] = vzero8();

  for (int k0 = 0; k0 < EMB; k0 += 32) {
    v16bf af[2], bf[4];
#pragma unroll
    for (int mt = 0; mt < 2; ++mt)
      af[mt] = load_a_frag(A + (size_t)(row0 + mt * 16 + l) * EMB + k0 + 8 * h);
#pragma unroll
    for (int nt = 0; nt < 4; ++nt)
      bf[nt] = load_a_frag(W + (size_t)(col0 + nt * 16 + l) * EMB + k0 + 8 * h);
#pragma unroll
    for (int mt = 0; mt < 2; ++mt)
#pragma unroll
      for (int nt = 0; nt < 4; ++nt)
        acc[mt][nt] = wmma_bf16(af[mt], bf[nt], acc[mt][nt]);
  }

#pragma unroll
  for (int mt = 0; mt < 2; ++mt) {
#pragma unroll
    for (int nt = 0; nt < 4; ++nt) {
      const int col = col0 + nt * 16 + l;
      const float bv = bias[col];
#pragma unroll
      for (int r = 0; r < 8; ++r) {
        const int row = row0 + mt * 16 + r + 8 * h;     // C layout: M = r + 8h
        const float v = (acc[mt][nt][r] + bv) * out_scale;
        if (MODE == 0) {
          ((float*)outp)[(size_t)row * EMB + col] = v;
        } else {
          const int bidx = row >> 11;                   // / NSEQ
          const int n    = row & (NSEQ - 1);
          const int head = col / DH;
          const int dk   = col - head * DH;
          if (MODE == 1)
            ((bf16_t*)outp)[((size_t)(bidx * NHEAD + head) * NSEQ + n) * DH + dk] = (bf16_t)v;
          else
            ((bf16_t*)outp)[((size_t)(bidx * NHEAD + head) * DH + dk) * NSEQ + n] = (bf16_t)v;
        }
      }
    }
  }
}

// ---------- flash attention: one wave owns 32 query rows (2 m_q tiles) ----------
// Q:[bh][N][d] (pre-scaled by log2e/sqrt(d)), K:[bh][N][d], Vt:[bh][d][N], Out bf16 [M][768]
// S^T = K * Q^T so the P^T accumulator layout directly feeds the O += P*V WMMA as A-operand.
// K fragments and V fragments are each shared by both q-tiles: 48 WMMAs per 24 fragment loads (48 b128) / iter.
// K fragments are double-buffered across j-tiles, V fragments use a 1-deep rotating prefetch,
// and waves_per_eu(1,2) keeps the allocator from serializing loads to chase occupancy.
__global__ __launch_bounds__(256) __attribute__((amdgpu_waves_per_eu(1, 2)))
void attn_kernel(const bf16_t* __restrict__ Q,
                 const bf16_t* __restrict__ Kb,
                 const bf16_t* __restrict__ Vt,
                 bf16_t* __restrict__ Out) {
  const int lane = threadIdx.x & 31;
  const int l = lane & 15, h = lane >> 4;
  const int wid = blockIdx.x * 8 + (threadIdx.x >> 5);   // 2048 waves total
  const int bh = wid >> 6;                               // head-batch index (0..31)
  const int q0 = (wid & 63) << 5;                        // 32-row query block

  const bf16_t* Qp = Q  + ((size_t)bh * NSEQ + q0) * DH;
  const bf16_t* Kp = Kb + (size_t)bh * NSEQ * DH;
  const bf16_t* Vp = Vt + (size_t)bh * DH * NSEQ;

  // Q^T as B-operand: lane = m_q (within q-tile), K = dk in table order (8*h offset, +16)
  v16bf qf[2][3];
#pragma unroll
  for (int qt = 0; qt < 2; ++qt)
#pragma unroll
    for (int kc = 0; kc < 3; ++kc)
      qf[qt][kc] = load_a_frag(Qp + (size_t)(qt * 16 + l) * DH + kc * 32 + 8 * h);

  v8f o[2][6];
#pragma unroll
  for (int qt = 0; qt < 2; ++qt)
#pragma unroll
    for (int nt = 0; nt < 6; ++nt) o[qt][nt] = vzero8();
  float mrun[2] = {-3.0e38f, -3.0e38f};   // per-lane: row m_q = qt*16 + (lane&15)
  float lrun[2] = {0.f, 0.f};

  for (int kv0 = 0; kv0 < NSEQ; kv0 += 64) {
    // S^T = K * Q^T : C layout => (kv = r + 8h + 16j, m_q = qt*16 + lane&15)
    v8f s[2][4];
#pragma unroll
    for (int qt = 0; qt < 2; ++qt)
#pragma unroll
      for (int j = 0; j < 4; ++j) s[qt][j] = vzero8();

    const bf16_t* kbase = Kp + ((size_t)kv0 + l) * DH + 8 * h;
    v16bf kf0[3], kf1[3];
#pragma unroll
    for (int kc = 0; kc < 3; ++kc) kf0[kc] = load_a_frag(kbase + kc * 32);
#pragma unroll
    for (int j = 0; j < 4; ++j) {
      v16bf (&cur)[3] = (j & 1) ? kf1 : kf0;
      v16bf (&nxt)[3] = (j & 1) ? kf0 : kf1;
      if (j < 3) {
#pragma unroll
        for (int kc = 0; kc < 3; ++kc)
          nxt[kc] = load_a_frag(kbase + (size_t)(j + 1) * 16 * DH + kc * 32);
      }
#pragma unroll
      for (int kc = 0; kc < 3; ++kc) {
        s[0][j] = wmma_bf16(cur[kc], qf[0][kc], s[0][j]);
        s[1][j] = wmma_bf16(cur[kc], qf[1][kc], s[1][j]);
      }
    }

    // online softmax row stats per q-tile (rows live on lanes; one xor-16 to merge halves)
    float corr[2];
#pragma unroll
    for (int qt = 0; qt < 2; ++qt) {
      float mx = mrun[qt];
#pragma unroll
      for (int j = 0; j < 4; ++j)
#pragma unroll
        for (int r = 0; r < 8; ++r) mx = fmaxf(mx, s[qt][j][r]);
      mx = fmaxf(mx, __shfl_xor(mx, 16, 32));
      corr[qt] = __builtin_amdgcn_exp2f(mrun[qt] - mx);
      mrun[qt] = mx;
      float rs = 0.f;
#pragma unroll
      for (int j = 0; j < 4; ++j)
#pragma unroll
        for (int r = 0; r < 8; ++r) {
          const float p = __builtin_amdgcn_exp2f(s[qt][j][r] - mx);
          s[qt][j][r] = p;
          rs += p;
        }
      rs += __shfl_xor(rs, 16, 32);
      lrun[qt] = lrun[qt] * corr[qt] + rs;
    }
    // rescale O (O rows = r + 8h -> broadcast corr from lane r+8h)
#pragma unroll
    for (int r = 0; r < 8; ++r) {
      const float cr0 = __shfl(corr[0], r + 8 * h, 32);
      const float cr1 = __shfl(corr[1], r + 8 * h, 32);
#pragma unroll
      for (int nt = 0; nt < 6; ++nt) {
        o[0][nt][r] *= cr0;
        o[1][nt][r] *= cr1;
      }
    }
    // O += P * V ; P^T accumulator layout IS the A-operand layout after bf16 pack.
    // V fragments shared by both q-tiles; rotating 1-deep prefetch across nt.
#pragma unroll
    for (int kc2 = 0; kc2 < 2; ++kc2) {
      v16bf pa[2];
#pragma unroll
      for (int qt = 0; qt < 2; ++qt) {
#pragma unroll
        for (int e = 0; e < 8; ++e) pa[qt][e]     = (bf16_t)s[qt][2 * kc2][e];
#pragma unroll
        for (int e = 0; e < 8; ++e) pa[qt][e + 8] = (bf16_t)s[qt][2 * kc2 + 1][e];
      }
      const bf16_t* vb = Vp + (size_t)l * NSEQ + kv0 + kc2 * 32 + 8 * h;
      v16bf vnext = load_a_frag(vb);
#pragma unroll
      for (int nt = 0; nt < 6; ++nt) {
        v16bf vcur = vnext;
        if (nt < 5) vnext = load_a_frag(vb + (size_t)(nt + 1) * 16 * NSEQ);
        o[0][nt] = wmma_bf16(pa[0], vcur, o[0][nt]);
        o[1][nt] = wmma_bf16(pa[1], vcur, o[1][nt]);
      }
    }
  }

  // normalize and store bf16 to [M][768]
  const int b = bh >> 3, head = bh & 7;
  bf16_t* op = Out + ((size_t)b * NSEQ + q0) * EMB + head * DH;
#pragma unroll
  for (int qt = 0; qt < 2; ++qt) {
    const float linv = 1.f / lrun[qt];
#pragma unroll
    for (int r = 0; r < 8; ++r) {
      const float li = __shfl(linv, r + 8 * h, 32);
#pragma unroll
      for (int nt = 0; nt < 6; ++nt)
        op[(size_t)(qt * 16 + r + 8 * h) * EMB + nt * 16 + l] = (bf16_t)(o[qt][nt][r] * li);
    }
  }
}

// ---------- host launcher ----------

extern "C" void kernel_launch(void* const* d_in, const int* in_sizes, int n_in,
                              void* d_out, int out_size, void* d_ws, size_t ws_size,
                              hipStream_t stream) {
  const float* x  = (const float*)d_in[0];
  const float* Wq = (const float*)d_in[1];
  const float* bq = (const float*)d_in[2];
  const float* Wk = (const float*)d_in[3];
  const float* bk = (const float*)d_in[4];
  const float* Wv = (const float*)d_in[5];
  const float* bv = (const float*)d_in[6];
  const float* Wo = (const float*)d_in[7];
  const float* bo = (const float*)d_in[8];

  char* ws = (char*)d_ws;
  size_t off = 0;
  auto alloc = [&](size_t bytes) {
    char* p = ws + off;
    off += (bytes + 255) & ~(size_t)255;
    return p;
  };
  bf16_t* xb  = (bf16_t*)alloc((size_t)MTOT * EMB * 2);
  bf16_t* Wqb = (bf16_t*)alloc((size_t)EMB * EMB * 2);
  bf16_t* Wkb = (bf16_t*)alloc((size_t)EMB * EMB * 2);
  bf16_t* Wvb = (bf16_t*)alloc((size_t)EMB * EMB * 2);
  bf16_t* Wob = (bf16_t*)alloc((size_t)EMB * EMB * 2);
  bf16_t* Qb  = (bf16_t*)alloc((size_t)MTOT * EMB * 2);
  bf16_t* Kbf = (bf16_t*)alloc((size_t)MTOT * EMB * 2);
  bf16_t* Vtb = (bf16_t*)alloc((size_t)MTOT * EMB * 2);
  bf16_t* Ab  = (bf16_t*)alloc((size_t)MTOT * EMB * 2);

  const int nx4 = MTOT * EMB / 4;
  const int nw4 = EMB * EMB / 4;
  cvt_kernel<<<(nx4 + 255) / 256, 256, 0, stream>>>(x,  xb,  nx4);
  cvt_kernel<<<(nw4 + 255) / 256, 256, 0, stream>>>(Wq, Wqb, nw4);
  cvt_kernel<<<(nw4 + 255) / 256, 256, 0, stream>>>(Wk, Wkb, nw4);
  cvt_kernel<<<(nw4 + 255) / 256, 256, 0, stream>>>(Wv, Wvb, nw4);
  cvt_kernel<<<(nw4 + 255) / 256, 256, 0, stream>>>(Wo, Wob, nw4);

  // fold 1/sqrt(d) * log2(e) into Q so the attention loop uses raw exp2
  const float qscale = 1.4426950408889634f / sqrtf((float)DH);

  dim3 grid(EMB / 128, MTOT / 128);
  gemm_kernel<1><<<grid, 256, 0, stream>>>(xb, Wqb, bq, Qb,  qscale);
  gemm_kernel<1><<<grid, 256, 0, stream>>>(xb, Wkb, bk, Kbf, 1.0f);
  gemm_kernel<2><<<grid, 256, 0, stream>>>(xb, Wvb, bv, Vtb, 1.0f);

  attn_kernel<<<256, 256, 0, stream>>>(Qb, Kbf, Vtb, Ab);

  gemm_kernel<0><<<grid, 256, 0, stream>>>(Ab, Wob, bo, d_out, 1.0f);
}
